// TransformerEncoder_27839978013166
// MI455X (gfx1250) — matrix operand order repaired and checked by reading, not compiled, not run on hardware
//
#include <hip/hip_runtime.h>
#include <hip/hip_fp16.h>
#include <cstdint>

typedef __attribute__((ext_vector_type(16))) _Float16 v16h;
typedef __attribute__((ext_vector_type(8)))  float    v8f;
typedef __attribute__((ext_vector_type(4)))  unsigned int u32x4;
typedef __attribute__((ext_vector_type(8)))  int      i32x8;
typedef __attribute__((ext_vector_type(4)))  int      i32x4;

#define DEVFN static __device__ __forceinline__
#define AS1 __attribute__((address_space(1)))
#define AS3 __attribute__((address_space(3)))

// ---- feature probes (fall back to sync paths when builtins are absent) ----
#if defined(__has_builtin)
#  if __has_builtin(__builtin_amdgcn_global_load_async_to_lds_b128)
#    define HAVE_ASYNC 1
#  endif
#  if __has_builtin(__builtin_amdgcn_tensor_load_to_lds)
#    define HAVE_TDM 1
#  endif
#endif
#ifndef HAVE_ASYNC
#  define HAVE_ASYNC 0
#endif
#ifndef HAVE_TDM
#  define HAVE_TDM 0
#endif

constexpr int FEATS = 1024;
constexpr int HEAD  = 16;
constexpr int HD    = 64;     // FEATS / HEAD
constexpr int MLPH  = 4096;
constexpr int BATCH = 8;
constexpr int SEQ   = 1024;
constexpr int ROWS  = BATCH * SEQ;   // 8192
constexpr float EPSLN = 1e-5f;

enum { EPI_QKV = 0, EPI_OPROJ = 1, EPI_MLP1 = 2, EPI_MLP2 = 3 };

// ---------------------------------------------------------------- helpers

DEVFN v8f wmma_f16(v16h a, v16h b, v8f c) {
  return __builtin_amdgcn_wmma_f32_16x16x32_f16(false, a, false, b, (short)0, c,
                                                false, false);
}

DEVFN float fq_act(float x, float alpha, float lvl) {
  float scale = alpha / lvl;
  x = fminf(fmaxf(x, -alpha), alpha);
  return rintf(x / scale) * scale;
}

DEVFN v16h load_afrag(const _Float16* s, int rbase, int stride, int lane) {
  int m  = lane & 15;
  int kb = (lane < 16) ? 0 : 16;
  const _Float16* p = s + (rbase + m) * stride + kb;
  v16h f;
#pragma unroll
  for (int i = 0; i < 8; ++i) { f[i] = p[i]; f[i + 8] = p[i + 8]; }
  return f;
}

DEVFN v16h load_bfrag(const _Float16* s, int nbase, int stride, int lane) {
  int n  = lane & 15;
  int kb = (lane < 16) ? 0 : 16;
  const _Float16* p = s + (nbase + n) * stride + kb;
  v16h f;
#pragma unroll
  for (int i = 0; i < 16; ++i) f[i] = p[i];
  return f;
}

DEVFN void wait_async0() {
#if defined(__has_builtin) && __has_builtin(__builtin_amdgcn_s_wait_asynccnt)
  __builtin_amdgcn_s_wait_asynccnt(0);
#else
  asm volatile("s_wait_asynccnt 0" ::: "memory");
#endif
}

#if HAVE_TDM
DEVFN unsigned lds_off_u32(const void* p) {
  return (unsigned)(uintptr_t)(AS3 const void*)p;
}

// 2D TDM load: tile of tile_m rows x tile_k halves, row stride = strideElems
// halves, from global tile start `g` into LDS byte offset `lds`.
DEVFN void tdm_load_tile(unsigned lds, const _Float16* g, int strideElems,
                         int numRows, int tile_k, int tile_m) {
  unsigned long long ga = (unsigned long long)(uintptr_t)g;
  u32x4 g0;
  g0[0] = 1u;                                   // count=1, user descriptor
  g0[1] = lds;                                  // lds_addr (bytes)
  g0[2] = (unsigned)ga;                         // global_addr[31:0]
  g0[3] = (unsigned)((ga >> 32) & 0x1FFFFFFu)   // global_addr[56:32]
          | (2u << 30);                         // type=2 (image)
  i32x8 g1;
  g1[0] = 1 << 16;                              // data_size=1 (2 bytes)
  g1[1] = (strideElems & 0xFFFF) << 16;         // tensor_dim0[15:0]
  g1[2] = (strideElems >> 16) | ((numRows & 0xFFFF) << 16);  // dim0 hi, dim1 lo
  g1[3] = (numRows >> 16) | (tile_k << 16);     // dim1 hi, tile_dim0
  g1[4] = tile_m;                               // tile_dim1 (tile_dim2 = 0)
  g1[5] = strideElems;                          // tensor_dim0_stride[31:0]
  g1[6] = 0;                                    // dim0_stride hi, dim1_stride lo
  g1[7] = 0;
  i32x4 z4 = {0, 0, 0, 0};
#if __clang_major__ >= 23
  i32x8 z8 = {0, 0, 0, 0, 0, 0, 0, 0};
  __builtin_amdgcn_tensor_load_to_lds(g0, g1, z4, z4, z8, 0);
#else
  __builtin_amdgcn_tensor_load_to_lds(g0, g1, z4, z4, 0);
#endif
}
#endif  // HAVE_TDM

// ---------------------------------------------------------------- small kernels

__global__ void k_zero4(float* a) {
  if (threadIdx.x < 4) a[threadIdx.x] = 0.0f;
}

__global__ __launch_bounds__(256) void k_absmax(const float* __restrict__ w,
                                                float* __restrict__ alpha, int n) {
  __shared__ float red[256];
  int tid = threadIdx.x;
  float m = 0.0f;
  for (int i = blockIdx.x * 256 + tid; i < n; i += gridDim.x * 256)
    m = fmaxf(m, fabsf(w[i]));
  red[tid] = m;
  __syncthreads();
  for (int o = 128; o > 0; o >>= 1) {
    if (tid < o) red[tid] = fmaxf(red[tid], red[tid + o]);
    __syncthreads();
  }
  if (tid == 0)
    atomicMax((unsigned int*)alpha, __float_as_uint(red[0]));
}

__global__ __launch_bounds__(256) void k_quant_w(const float* __restrict__ w,
                                                 const float* __restrict__ alphaP,
                                                 _Float16* __restrict__ o, int n) {
  int i = blockIdx.x * 256 + threadIdx.x;
  if (i >= n) return;
  float alpha = alphaP[0];
  float scale = alpha / 7.0f;
  float q = rintf(fminf(fmaxf(w[i] / scale, -7.0f), 7.0f)) * scale;
  o[i] = (_Float16)q;
}

__global__ __launch_bounds__(256) void k_cvt(const float* __restrict__ w,
                                             _Float16* __restrict__ o, int n) {
  int i = blockIdx.x * 256 + threadIdx.x;
  if (i < n) o[i] = (_Float16)w[i];
}

__global__ __launch_bounds__(256) void k_ln(const float* __restrict__ in,
                                            const float* __restrict__ g,
                                            const float* __restrict__ bb,
                                            _Float16* __restrict__ out,
                                            int doQuant, float qAlpha) {
  __shared__ float red[256];
  int r = blockIdx.x, tid = threadIdx.x;
  const float* row = in + (size_t)r * FEATS;
  float x[4], s = 0.0f;
#pragma unroll
  for (int i = 0; i < 4; ++i) { x[i] = row[tid + i * 256]; s += x[i]; }
  red[tid] = s; __syncthreads();
  for (int o = 128; o > 0; o >>= 1) { if (tid < o) red[tid] += red[tid + o]; __syncthreads(); }
  float mean = red[0] * (1.0f / FEATS);
  __syncthreads();
  s = 0.0f;
#pragma unroll
  for (int i = 0; i < 4; ++i) { float d = x[i] - mean; s += d * d; }
  red[tid] = s; __syncthreads();
  for (int o = 128; o > 0; o >>= 1) { if (tid < o) red[tid] += red[tid + o]; __syncthreads(); }
  float rstd = rsqrtf(red[0] * (1.0f / FEATS) + EPSLN);
#pragma unroll
  for (int i = 0; i < 4; ++i) {
    int c = tid + i * 256;
    float y = (x[i] - mean) * rstd * g[c] + bb[c];
    if (doQuant) y = fq_act(y, qAlpha, 7.0f);
    out[(size_t)r * FEATS + c] = (_Float16)y;
  }
}

// ---------------------------------------------------------------- fused GEMM
// C[M,N] = A[M,K](f16,row-major) * Bt[N,K](f16) + epilogue.
// Block tile 128x128, 8 waves (4 along M x 2 along N), wave tile 32x64.
// Tiles staged by the Tensor Data Mover (double-buffered) when available.
template <int EPI>
__global__ __launch_bounds__(256) void k_gemm(const _Float16* __restrict__ A,
                                              const _Float16* __restrict__ Bt,
                                              const float* __restrict__ bias,
                                              const float* __restrict__ resid,
                                              float* __restrict__ outF,
                                              _Float16* __restrict__ outH,
                                              int M, int N, int K, float qAlpha) {
  __shared__ __align__(16) _Float16 sA[2][128 * 32];
  __shared__ __align__(16) _Float16 sB[2][128 * 32];

  int tid = threadIdx.x, lane = tid & 31, wave = tid >> 5;
  int wm = wave & 3, wn = wave >> 2;
  int m0 = blockIdx.y * 128, n0 = blockIdx.x * 128;

  v8f acc[2][4];
#pragma unroll
  for (int tm = 0; tm < 2; ++tm)
#pragma unroll
    for (int tn = 0; tn < 4; ++tn)
#pragma unroll
      for (int i = 0; i < 8; ++i) acc[tm][tn][i] = 0.0f;

  const int nT = K >> 5;

#if HAVE_TDM
  // --- TDM double-buffered staging: wave0 drives the Tensor Data Mover ---
  if (tid == 0) {
    tdm_load_tile(lds_off_u32(&sA[0][0]), A + (size_t)m0 * K, K, M, 32, 128);
    tdm_load_tile(lds_off_u32(&sB[0][0]), Bt + (size_t)n0 * K, K, N, 32, 128);
  }
  for (int t = 0; t < nT; ++t) {
    int cur = t & 1;
    if (tid == 0) {
      if (t + 1 < nT) {
        tdm_load_tile(lds_off_u32(&sA[cur ^ 1][0]),
                      A + (size_t)m0 * K + (t + 1) * 32, K, M, 32, 128);
        tdm_load_tile(lds_off_u32(&sB[cur ^ 1][0]),
                      Bt + (size_t)n0 * K + (t + 1) * 32, K, N, 32, 128);
        __builtin_amdgcn_s_wait_tensorcnt(2);   // previous pair complete
      } else {
        __builtin_amdgcn_s_wait_tensorcnt(0);   // final tile complete
      }
    }
    __syncthreads();
    const _Float16* pA = sA[cur];
    const _Float16* pB = sB[cur];
    v16h a0 = load_afrag(pA, wm * 32,      32, lane);
    v16h a1 = load_afrag(pA, wm * 32 + 16, 32, lane);
#pragma unroll
    for (int tn = 0; tn < 4; ++tn) {
      v16h b = load_bfrag(pB, wn * 64 + tn * 16, 32, lane);
      acc[0][tn] = wmma_f16(a0, b, acc[0][tn]);
      acc[1][tn] = wmma_f16(a1, b, acc[1][tn]);
    }
    __syncthreads();
  }
#else
  // --- synchronous fallback staging ---
  int lrow = tid >> 1, loff = (tid & 1) * 16;
  for (int t = 0; t < nT; ++t) {
    int k0 = t * 32;
    const uint4* ga = (const uint4*)(A  + (size_t)(m0 + lrow) * K + k0 + loff);
    const uint4* gb = (const uint4*)(Bt + (size_t)(n0 + lrow) * K + k0 + loff);
    uint4* da = (uint4*)(&sA[0][lrow * 32 + loff]);
    uint4* db = (uint4*)(&sB[0][lrow * 32 + loff]);
    da[0] = ga[0]; da[1] = ga[1];
    db[0] = gb[0]; db[1] = gb[1];
    if (k0 + 32 < K) {
      __builtin_prefetch(A  + (size_t)(m0 + lrow) * K + k0 + 32 + loff, 0, 0);
      __builtin_prefetch(Bt + (size_t)(n0 + lrow) * K + k0 + 32 + loff, 0, 0);
    }
    __syncthreads();
    v16h a0 = load_afrag(sA[0], wm * 32,      32, lane);
    v16h a1 = load_afrag(sA[0], wm * 32 + 16, 32, lane);
#pragma unroll
    for (int tn = 0; tn < 4; ++tn) {
      v16h b = load_bfrag(sB[0], wn * 64 + tn * 16, 32, lane);
      acc[0][tn] = wmma_f16(a0, b, acc[0][tn]);
      acc[1][tn] = wmma_f16(a1, b, acc[1][tn]);
    }
    __syncthreads();
  }
#endif

  // epilogue: C layout -> VGPR v holds row v (+8 for lanes 16..31), col = lane&15
#pragma unroll
  for (int tm = 0; tm < 2; ++tm) {
#pragma unroll
    for (int tn = 0; tn < 4; ++tn) {
      int r0 = m0 + wm * 32 + tm * 16 + ((lane < 16) ? 0 : 8);
      int c  = n0 + wn * 64 + tn * 16 + (lane & 15);
      float bcol = bias[c];
#pragma unroll
      for (int vv = 0; vv < 8; ++vv) {
        int r = r0 + vv;
        float val = acc[tm][tn][vv] + bcol;
        if constexpr (EPI == EPI_QKV) {
          val = fq_act(val, qAlpha, 7.0f);
          int bb_ = r >> 10, nn_ = r & 1023, hh_ = c >> 6, dd_ = c & 63;
          outH[(((size_t)bb_ * HEAD + hh_) * SEQ + nn_) * HD + dd_] = (_Float16)val;
        } else if constexpr (EPI == EPI_OPROJ) {
          size_t idx = (size_t)r * N + c;
          outF[idx] = val + resid[idx];
        } else if constexpr (EPI == EPI_MLP1) {
          outH[(size_t)r * N + c] = (_Float16)fmaxf(val, 0.0f);
        } else {  // EPI_MLP2
          size_t idx = (size_t)r * N + c;
          outF[idx] = fmaxf(val, 0.0f) + resid[idx];
        }
      }
    }
  }
}

// ---------------------------------------------------------------- flash attention
// grid (B*H, SEQ/128); 8 waves, each owns 16 query rows. K/V tiles staged to
// LDS with async loads (ASYNCcnt) when available; online softmax; P bounced
// through wave-private LDS (C-layout -> A-layout) with s_wait_dscnt.
__global__ __launch_bounds__(256) void k_attn(const _Float16* __restrict__ q,
                                              const _Float16* __restrict__ k,
                                              const _Float16* __restrict__ v,
                                              _Float16* __restrict__ attn,
                                              const float* __restrict__ deqP,
                                              const float* __restrict__ vdeqP) {
  __shared__ __align__(16) _Float16 sK[128 * HD];
  __shared__ __align__(16) _Float16 sV[128 * HD];
  __shared__ __align__(16) _Float16 sP[8 * 16 * 32];

  int bh = blockIdx.x, qt = blockIdx.y;
  int tid = threadIdx.x, lane = tid & 31, wave = tid >> 5;
  float deq = deqP[0], vdeq = vdeqP[0];

  const _Float16* qbase = q + (size_t)bh * SEQ * HD + ((size_t)qt * 128 + wave * 16) * HD;
  v16h aq0, aq1;
  {
    int m = lane & 15, kb = (lane < 16) ? 0 : 16;
    const _Float16* p = qbase + m * HD + kb;
#pragma unroll
    for (int i = 0; i < 8; ++i) {
      aq0[i] = p[i];      aq0[i + 8] = p[i + 8];
      aq1[i] = p[i + 32]; aq1[i + 8] = p[i + 40];
    }
  }

  float mi[8], li[8];
  v8f O[4];
#pragma unroll
  for (int i = 0; i < 8; ++i) { mi[i] = -INFINITY; li[i] = 0.0f; }
#pragma unroll
  for (int t = 0; t < 4; ++t)
#pragma unroll
    for (int i = 0; i < 8; ++i) O[t][i] = 0.0f;

  const _Float16* kg = k + (size_t)bh * SEQ * HD;
  const _Float16* vg = v + (size_t)bh * SEQ * HD;
  _Float16* sPw = sP + wave * 16 * 32;

  for (int jt = 0; jt < SEQ; jt += 128) {
    __syncthreads();
#if HAVE_ASYNC
    {
      const char* gk = (const char*)(kg + (size_t)jt * HD);
      const char* gv = (const char*)(vg + (size_t)jt * HD);
      char* dk = (char*)sK;
      char* dv = (char*)sV;
#pragma unroll
      for (int i = 0; i < 4; ++i) {
        int o16 = (tid + 256 * i) * 16;
        __builtin_amdgcn_global_load_async_to_lds_b128(
            (AS1 i32x4*)(gk + o16), (AS3 i32x4*)(dk + o16), 0, 0);
        __builtin_amdgcn_global_load_async_to_lds_b128(
            (AS1 i32x4*)(gv + o16), (AS3 i32x4*)(dv + o16), 0, 0);
      }
      wait_async0();
    }
#else
    {
      const uint4* gk = (const uint4*)(kg + (size_t)jt * HD);
      const uint4* gv = (const uint4*)(vg + (size_t)jt * HD);
      uint4* dk = (uint4*)sK; uint4* dv = (uint4*)sV;
#pragma unroll
      for (int i = 0; i < 4; ++i) {
        dk[tid + 256 * i] = gk[tid + 256 * i];
        dv[tid + 256 * i] = gv[tid + 256 * i];
      }
    }
#endif
    __syncthreads();

    for (int j = 0; j < 128; j += 32) {
      v8f s0, s1;
#pragma unroll
      for (int i = 0; i < 8; ++i) { s0[i] = 0.0f; s1[i] = 0.0f; }
      {
        v16h bk;
        bk = load_bfrag(sK, j, HD, lane);            s0 = wmma_f16(aq0, bk, s0);
        {
          int n = lane & 15, kb = (lane < 16) ? 0 : 16;
          const _Float16* p = sK + (j + n) * HD + 32 + kb;
#pragma unroll
          for (int i = 0; i < 16; ++i) bk[i] = p[i];
        }
        s0 = wmma_f16(aq1, bk, s0);
        bk = load_bfrag(sK, j + 16, HD, lane);       s1 = wmma_f16(aq0, bk, s1);
        {
          int n = lane & 15, kb = (lane < 16) ? 0 : 16;
          const _Float16* p = sK + (j + 16 + n) * HD + 32 + kb;
#pragma unroll
          for (int i = 0; i < 16; ++i) bk[i] = p[i];
        }
        s1 = wmma_f16(aq1, bk, s1);
      }

      float fs[8];
#pragma unroll
      for (int vv = 0; vv < 8; ++vv) {
        float a = s0[vv] * deq, b2 = s1[vv] * deq;
        float mx = fmaxf(a, b2);
#pragma unroll
        for (int off = 8; off > 0; off >>= 1) mx = fmaxf(mx, __shfl_xor(mx, off, 16));
        float mnew = fmaxf(mi[vv], mx);
        float p0 = __expf(a - mnew), p1 = __expf(b2 - mnew);
        float f = __expf(mi[vv] - mnew);
        float sum = p0 + p1;
#pragma unroll
        for (int off = 8; off > 0; off >>= 1) sum += __shfl_xor(sum, off, 16);
        li[vv] = li[vv] * f + sum;
        mi[vv] = mnew;
        fs[vv] = f;
        int rr = vv + ((lane < 16) ? 0 : 8);
        sPw[rr * 32 + (lane & 15)]      = (_Float16)p0;
        sPw[rr * 32 + 16 + (lane & 15)] = (_Float16)p1;
      }
#pragma unroll
      for (int t = 0; t < 4; ++t)
#pragma unroll
        for (int vv = 0; vv < 8; ++vv) O[t][vv] *= fs[vv];

      // wave-private LDS bounce: wait only on our own DS stores
      asm volatile("s_wait_dscnt 0" ::: "memory");
      v16h ap = load_afrag(sPw, 0, 32, lane);

#pragma unroll
      for (int t = 0; t < 4; ++t) {
        v16h bv;
        int n = lane & 15, kb = (lane < 16) ? 0 : 16;
#pragma unroll
        for (int i = 0; i < 16; ++i) bv[i] = sV[(j + kb + i) * HD + t * 16 + n];
        O[t] = wmma_f16(ap, bv, O[t]);
      }
    }
  }

  int b = bh / HEAD, h = bh % HEAD;
#pragma unroll
  for (int vv = 0; vv < 8; ++vv) {
    float inv = vdeq / li[vv];
    int n = qt * 128 + wave * 16 + vv + ((lane < 16) ? 0 : 8);
    size_t base = ((size_t)(b * SEQ + n)) * FEATS + h * HD + (lane & 15);
#pragma unroll
    for (int t = 0; t < 4; ++t) attn[base + t * 16] = (_Float16)(O[t][vv] * inv);
  }
}

// ---------------------------------------------------------------- launcher

extern "C" void kernel_launch(void* const* d_in, const int* in_sizes, int n_in,
                              void* d_out, int out_size, void* d_ws, size_t ws_size,
                              hipStream_t stream) {
  const float* x    = (const float*)d_in[0];
  const float* ln1g = (const float*)d_in[1];
  const float* ln1b = (const float*)d_in[2];
  const float* w_q  = (const float*)d_in[3];
  const float* b_q  = (const float*)d_in[4];
  const float* w_k  = (const float*)d_in[5];
  const float* b_k  = (const float*)d_in[6];
  const float* w_v  = (const float*)d_in[7];
  const float* b_v  = (const float*)d_in[8];
  const float* w_o  = (const float*)d_in[9];
  const float* b_o  = (const float*)d_in[10];
  const float* ln2g = (const float*)d_in[11];
  const float* ln2b = (const float*)d_in[12];
  const float* w1   = (const float*)d_in[13];
  const float* b1   = (const float*)d_in[14];
  const float* w2   = (const float*)d_in[15];
  const float* b2   = (const float*)d_in[16];
  const float* deq  = (const float*)d_in[17];
  const float* vdeq = (const float*)d_in[18];
  float* out = (float*)d_out;

  char* ws = (char*)d_ws;
  size_t off = 0;
  auto alloc = [&](size_t bytes) -> void* {
    void* p = ws + off;
    off = (off + bytes + 255) & ~(size_t)255;
    return p;
  };
  float*    alphas = (float*)alloc(4 * sizeof(float));
  _Float16* xq     = (_Float16*)alloc((size_t)ROWS * FEATS * 2);   // reused as xn2
  _Float16* wq16   = (_Float16*)alloc((size_t)FEATS * FEATS * 2);
  _Float16* wk16   = (_Float16*)alloc((size_t)FEATS * FEATS * 2);
  _Float16* wv16   = (_Float16*)alloc((size_t)FEATS * FEATS * 2);
  _Float16* wo16   = (_Float16*)alloc((size_t)FEATS * FEATS * 2);
  _Float16* w1h    = (_Float16*)alloc((size_t)MLPH * FEATS * 2);
  _Float16* w2h    = (_Float16*)alloc((size_t)FEATS * MLPH * 2);
  _Float16* qb     = (_Float16*)alloc((size_t)ROWS * FEATS * 2);   // [b,h,n,d]
  _Float16* kb     = (_Float16*)alloc((size_t)ROWS * FEATS * 2);
  _Float16* vb     = (_Float16*)alloc((size_t)ROWS * FEATS * 2);
  _Float16* attn   = (_Float16*)alloc((size_t)ROWS * FEATS * 2);
  float*    out1   = (float*)alloc((size_t)ROWS * FEATS * 4);
  _Float16* xn2    = xq;                 // xq dead after QKV projections
  _Float16* hbuf   = qb;                 // q/k/v/attn (64MB) dead after o-proj

  const int WN = FEATS * FEATS;  // 1M

  k_zero4<<<1, 64, 0, stream>>>(alphas);
  k_absmax<<<64, 256, 0, stream>>>(w_q, alphas + 0, WN);
  k_absmax<<<64, 256, 0, stream>>>(w_k, alphas + 1, WN);
  k_absmax<<<64, 256, 0, stream>>>(w_v, alphas + 2, WN);
  k_absmax<<<64, 256, 0, stream>>>(w_o, alphas + 3, WN);
  k_quant_w<<<WN / 256, 256, 0, stream>>>(w_q, alphas + 0, wq16, WN);
  k_quant_w<<<WN / 256, 256, 0, stream>>>(w_k, alphas + 1, wk16, WN);
  k_quant_w<<<WN / 256, 256, 0, stream>>>(w_v, alphas + 2, wv16, WN);
  k_quant_w<<<WN / 256, 256, 0, stream>>>(w_o, alphas + 3, wo16, WN);
  k_cvt<<<(MLPH * FEATS) / 256, 256, 0, stream>>>(w1, w1h, MLPH * FEATS);
  k_cvt<<<(FEATS * MLPH) / 256, 256, 0, stream>>>(w2, w2h, FEATS * MLPH);

  k_ln<<<ROWS, 256, 0, stream>>>(x, ln1g, ln1b, xq, 1, 8.0f);

  dim3 gQ(FEATS / 128, ROWS / 128);
  k_gemm<EPI_QKV><<<gQ, 256, 0, stream>>>(xq, wq16, b_q, nullptr, nullptr, qb,
                                          ROWS, FEATS, FEATS, 2.0f);
  k_gemm<EPI_QKV><<<gQ, 256, 0, stream>>>(xq, wk16, b_k, nullptr, nullptr, kb,
                                          ROWS, FEATS, FEATS, 2.0f);
  k_gemm<EPI_QKV><<<gQ, 256, 0, stream>>>(xq, wv16, b_v, nullptr, nullptr, vb,
                                          ROWS, FEATS, FEATS, 2.0f);

  k_attn<<<dim3(BATCH * HEAD, SEQ / 128), 256, 0, stream>>>(qb, kb, vb, attn,
                                                            deq, vdeq);

  k_gemm<EPI_OPROJ><<<gQ, 256, 0, stream>>>(attn, wo16, b_o, x, out1, nullptr,
                                            ROWS, FEATS, FEATS, 0.0f);

  k_ln<<<ROWS, 256, 0, stream>>>(out1, ln2g, ln2b, xn2, 0, 0.0f);
  k_gemm<EPI_MLP1><<<dim3(MLPH / 128, ROWS / 128), 256, 0, stream>>>(
      xn2, w1h, b1, nullptr, nullptr, hbuf, ROWS, MLPH, FEATS, 0.0f);
  k_gemm<EPI_MLP2><<<dim3(FEATS / 128, ROWS / 128), 256, 0, stream>>>(
      hbuf, w2h, b2, out1, out, nullptr, ROWS, FEATS, MLPH, 0.0f);
}
